// SpatioTemporalLossWithFloatingCentre_36610301231306
// MI455X (gfx1250) — matrix operand order repaired and checked by reading, not compiled, not run on hardware
//
#include <hip/hip_runtime.h>
#include <hip/hip_bf16.h>
#include <math.h>

// ---------------------------------------------------------------------------
// SpatioTemporalLossWithFloatingCentre for MI455X (gfx1250, wave32)
// Pipeline:
//   k_zero      : zero stats header in workspace
//   k_stats     : fused tanh/coords/sigma feature + mask staging into LDS
//                 (row-major per feature/mask), then v_wmma_f32_16x16x32_f16
//                 masked reduction; inner loop = 4x ds_load_b128 + 1x v_wmma
//                 per (b,k,t): [Σm·se0, Σm·se1, Σm·se2, Σm·σ, Σm·σ², cnt]
//   k_fin1/2    : s_kt, exp(10 s), var terms, scalar centre per (b,k)
//   per (b,k) slice (6x):
//     k_dist    : dist/err + seed-loss partials + pack (key|label) u64 records
//     4x radix pass (k_hist -> k_scan -> k_scatter), stable & deterministic
//     k_lblsum / k_lblscan / k_lovasz : label cumsum + Lovász grad dot
//   k_final     : combine  W_INST*inst + W_VAR*var + W_SEED*seed, mean over b
// ---------------------------------------------------------------------------

typedef __attribute__((ext_vector_type(16))) _Float16 v16h;
typedef __attribute__((ext_vector_type(8)))  _Float16 v8h;
typedef __attribute__((ext_vector_type(8)))  float    v8f;

#define B_   2
#define K_   3
#define T_   16
#define H_   320
#define W_   640
#define HW_  (H_*W_)
#define THW_ (T_*HW_)

#define SX_ (4.16f/1999.f)
#define SY_ (1.6f/799.f)
#define ST_ (0.1f/31.f)

// header layout (floats) inside workspace
#define OFF_ACC     0      // [B][T][K][6]  = 576
#define OFF_CENTRE  576    // [B][K]        = 6   (scalar centre per instance)
#define OFF_VAR     582    // [B]           = 2
#define OFF_SEEDFG  584    // [B]
#define OFF_SEEDBG  586    // [B]
#define OFF_INSTL   588    // [B][K]        = 6
#define OFF_SEXP    594    // [B][K][T]     = 96
#define HDR_FLOATS  1024

#define RTILE 2048                   // elements per block for sort-side kernels
#define NB_   (THW_ / RTILE)         // 1600 blocks over one (b,k) slice

__device__ __forceinline__ unsigned ballot32(bool p) {
#if __has_builtin(__builtin_amdgcn_ballot_w32)
  return __builtin_amdgcn_ballot_w32(p);
#else
  return (unsigned)__ballot(p);
#endif
}

// ---------------------------------------------------------------- k_zero ---
__global__ void k_zero(float* hdr) {
  int i = blockIdx.x * 256 + threadIdx.x;
  if (i < HDR_FLOATS) hdr[i] = 0.0f;
}

// --------------------------------------------------------------- k_stats ---
// grid = B*T*100 blocks of 256 threads; each block: one 2048-pixel tile.
// Stage f16 features (feature-major) and f16 instance masks (row-major, row 3
// = zeros for unused A rows) to LDS; each of the 8 waves reduces its 256-pixel
// segment with 8 unrolled WMMA steps (K=32 pixels each), then atomically
// accumulates the 3x6 useful block of the 16x16 f32 accumulator.
// Lanes with column n>=8 read feature row n&7: their products only touch C
// columns 8..15 which are never read; A rows m>=3 read the zero mask row.
// Inner loop is branch-free: 4x ds_load_b128 + 1x v_wmma per step.
__global__ void k_stats(const float* __restrict__ pred,
                        const int*  __restrict__ inst,
                        float* __restrict__ hdr) {
  __shared__ __align__(16) _Float16 featT[8 * RTILE];  // [feature][pixel], 32 KB
  __shared__ __align__(16) _Float16 maskT[4 * RTILE];  // [row][pixel],     16 KB

  const int ti    = threadIdx.x;
  const int chunk = blockIdx.x % 100;
  const int bt    = blockIdx.x / 100;
  const int b     = bt / T_;
  const int t     = bt % T_;
  const int pbase = chunk * RTILE;

  const float* p0 = pred + ((b*5 + 0)*T_ + t) * HW_;
  const float* p1 = pred + ((b*5 + 1)*T_ + t) * HW_;
  const float* p2 = pred + ((b*5 + 2)*T_ + t) * HW_;
  const float* p3 = pred + ((b*5 + 3)*T_ + t) * HW_;
  const int*   ib = inst + (b*T_ + t) * HW_;
  const float  tc = (float)t * ST_;

  for (int j = 0; j < 8; ++j) {
    const int pl = j*256 + ti;
    const int p  = pbase + pl;
    __builtin_prefetch(p0 + p + RTILE, 0, 1);   // global_prefetch_b8
    __builtin_prefetch(p3 + p + RTILE, 0, 1);
    const float a0 = p0[p], a1 = p1[p], a2 = p2[p], sg = p3[p];
    const int row = p / W_, col = p % W_;
    const float e0 = tanhf(a0) + tc;
    const float e1 = tanhf(a1) + (float)row * SY_;
    const float e2 = tanhf(a2) + (float)col * SX_;
    featT[0*RTILE + pl] = (_Float16)e0;
    featT[1*RTILE + pl] = (_Float16)e1;
    featT[2*RTILE + pl] = (_Float16)e2;
    featT[3*RTILE + pl] = (_Float16)sg;
    featT[4*RTILE + pl] = (_Float16)(sg*sg);
    featT[5*RTILE + pl] = (_Float16)1.0f;
    featT[6*RTILE + pl] = (_Float16)0.0f;
    featT[7*RTILE + pl] = (_Float16)0.0f;
    const int id = ib[p];
    maskT[0*RTILE + pl] = (id == 1) ? (_Float16)1.0f : (_Float16)0.0f;
    maskT[1*RTILE + pl] = (id == 2) ? (_Float16)1.0f : (_Float16)0.0f;
    maskT[2*RTILE + pl] = (id == 3) ? (_Float16)1.0f : (_Float16)0.0f;
    maskT[3*RTILE + pl] = (_Float16)0.0f;
  }
  __syncthreads();

  const int lane = ti & 31, wv = ti >> 5, half = lane >> 4;
  const int m  = lane & 15;             // A row this lane owns
  const int n  = lane & 15;             // C column this lane owns
  const int nn = lane & 7;              // feature row (n>=8 -> unused cols)
  const int mm = (m < 3) ? m : 3;       // mask row (m>=3 -> zero row)
  const int seg = wv * 256;

  const v8h* fa = (const v8h*)maskT;
  const v8h* fb = (const v8h*)featT;

  v8f c = {};
#pragma unroll
  for (int step = 0; step < 8; ++step) {
    const int pb0 = seg + step*32;
    // A fragment: kk pattern {0..7,16..23}+8*half -> two ds_load_b128
    union { v16h v; v8h h[2]; } au;
    au.h[0] = fa[(mm*RTILE + pb0 + 8*half) >> 3];
    au.h[1] = fa[(mm*RTILE + pb0 + 16 + 8*half) >> 3];
    // B fragment: kk pattern {0..7,16..23}+8*half of feature row nn -> two ds_load_b128
    union { v16h v; v8h h[2]; } bu;
    bu.h[0] = fb[(nn*RTILE + pb0 + 8*half) >> 3];
    bu.h[1] = fb[(nn*RTILE + pb0 + 16 + 8*half) >> 3];
    c = __builtin_amdgcn_wmma_f32_16x16x32_f16(false, au.v, false, bu.v,
                                               (short)0, c, false, false);
  }
  // C layout: VGPR r -> M = r + 8*half, N = lane%16
  if (half == 0 && n < 6) {
    for (int r = 0; r < 3; ++r)
      atomicAdd(&hdr[OFF_ACC + ((b*T_ + t)*K_ + r)*6 + n], c[r]);
  }
}

// ---------------------------------------------------------------- k_fin1 ---
__global__ void k_fin1(float* hdr) {
  const int tid = threadIdx.x;
  if (tid >= B_*K_*T_) return;
  const int b = tid / (K_*T_), k = (tid / T_) % K_, t = tid % T_;
  const float* a = &hdr[OFF_ACC + ((b*T_ + t)*K_ + k)*6];
  float cnt = a[5]; if (cnt < 0.5f) cnt = 1.0f;
  const float s   = a[3] / cnt;
  const float var = a[4] / cnt - s*s;           // Σmσ²/cnt − s̄²
  hdr[OFF_SEXP + (b*K_ + k)*T_ + t] = expf(10.0f * s);
  atomicAdd(&hdr[OFF_CENTRE + b*K_ + k], (a[0] + a[1] + a[2]) / cnt);
  atomicAdd(&hdr[OFF_VAR + b], var);
}

__global__ void k_fin2(float* hdr) {
  const int tid = threadIdx.x;
  if (tid < B_*K_) hdr[OFF_CENTRE + tid] *= (1.0f / (float)(T_*3));
}

// ---------------------------------------------------------------- k_dist ---
// per (b,k) slice: dist, hinge error -> packed u64 record, seed-loss partials
__global__ void k_dist(const float* __restrict__ pred,
                       const int*  __restrict__ inst,
                       float* __restrict__ hdr,
                       unsigned long long* __restrict__ recs,
                       int b, int k, int doBg) {
  __shared__ float redFg[256];
  __shared__ float redBg[256];
  const int ti = threadIdx.x;
  const float centre = hdr[OFF_CENTRE + b*K_ + k];
  float fg = 0.0f, bg = 0.0f;

  for (int j = 0; j < 4; ++j) {
    const int idx = blockIdx.x*1024 + j*256 + ti;
    const int t = idx / HW_, p = idx % HW_;
    const int row = p / W_, col = p % W_;
    const float* pb = pred + (b*5*T_ + t) * HW_ + p;
    __builtin_prefetch(pb + 1024, 0, 1);
    const float a0 = pb[0*T_*HW_], a1 = pb[1*T_*HW_];
    const float a2 = pb[2*T_*HW_], a4 = pb[4*T_*HW_];
    const float e0 = tanhf(a0) + (float)t  * ST_ - centre;
    const float e1 = tanhf(a1) + (float)row* SY_ - centre;
    const float e2 = tanhf(a2) + (float)col* SX_ - centre;
    const float d2   = e0*e0 + e1*e1 + e2*e2;
    const float sexp = hdr[OFF_SEXP + (b*K_ + k)*T_ + t];
    const float dist = expf(-d2 * sexp);
    const int   id   = inst[(b*T_ + t)*HW_ + p];
    const int   lbl  = (id == k + 1) ? 1 : 0;
    const float sign = lbl ? 1.0f : -1.0f;
    const float err  = 1.0f - (2.0f*dist - 1.0f) * sign;
    // descending sort of err == ascending sort of kd = ~monotonic(err)
    const unsigned u  = __float_as_uint(err);
    const unsigned ka = (u & 0x80000000u) ? ~u : (u | 0x80000000u);
    const unsigned kd = ~ka;
    recs[idx] = ((unsigned long long)kd << 32) | (unsigned)lbl;
    const float seed = 1.0f / (1.0f + expf(-a4));
    if (lbl) { const float dd = seed - dist; fg += dd*dd; }
    if (doBg && id == 0) bg += seed*seed;
  }
  redFg[ti] = fg; redBg[ti] = bg; __syncthreads();
  for (int s = 128; s > 0; s >>= 1) {
    if (ti < s) { redFg[ti] += redFg[ti+s]; redBg[ti] += redBg[ti+s]; }
    __syncthreads();
  }
  if (ti == 0) {
    atomicAdd(&hdr[OFF_SEEDFG + b], redFg[0]);
    if (doBg) atomicAdd(&hdr[OFF_SEEDBG + b], redBg[0]);
  }
}

// ------------------------------------------------------------ radix sort ---
__global__ void k_hist(const unsigned long long* __restrict__ src,
                       unsigned* __restrict__ hist, int shift) {
  __shared__ unsigned cnt[256];
  const int ti = threadIdx.x;
  cnt[ti] = 0; __syncthreads();
  for (int j = 0; j < 8; ++j) {
    const unsigned d =
      (unsigned)((src[(size_t)blockIdx.x*RTILE + j*256 + ti] >> shift) & 255ull);
    atomicAdd(&cnt[d], 1u);
  }
  __syncthreads();
  hist[(size_t)ti*NB_ + blockIdx.x] = cnt[ti];
}

__global__ void k_scan(const unsigned* __restrict__ hist,
                       unsigned* __restrict__ offs) {
  __shared__ unsigned totals[256];
  __shared__ unsigned dbase[256];
  const int d = threadIdx.x;
  unsigned s = 0;
  for (int i = 0; i < NB_; ++i) s += hist[(size_t)d*NB_ + i];
  totals[d] = s; __syncthreads();
  if (d == 0) { unsigned r = 0; for (int i = 0; i < 256; ++i) { dbase[i] = r; r += totals[i]; } }
  __syncthreads();
  unsigned run = dbase[d];
  for (int i = 0; i < NB_; ++i) {
    const unsigned t = hist[(size_t)d*NB_ + i];
    offs[(size_t)d*NB_ + i] = run; run += t;
  }
}

// stable, deterministic scatter: ballot multisplit, no atomic ranking
__global__ void k_scatter(const unsigned long long* __restrict__ src,
                          unsigned long long* __restrict__ dst,
                          const unsigned* __restrict__ offs, int shift) {
  __shared__ unsigned waveHist[8][256];
  __shared__ unsigned runCnt[256];
  const int ti = threadIdx.x, lane = ti & 31, wv = ti >> 5;
  runCnt[ti] = 0;
  __syncthreads();
  for (int j = 0; j < 8; ++j) {
    for (int w = 0; w < 8; ++w) waveHist[w][ti] = 0;
    __syncthreads();
    const unsigned long long rec = src[(size_t)blockIdx.x*RTILE + j*256 + ti];
    const unsigned d = (unsigned)((rec >> shift) & 255ull);
    unsigned mask = 0xFFFFFFFFu;
    for (int bb = 0; bb < 8; ++bb) {
      const unsigned bal = ballot32(((d >> bb) & 1u) != 0u);
      mask &= ((d >> bb) & 1u) ? bal : ~bal;
    }
    const unsigned laneRank = __popc(mask & ((1u << lane) - 1u));
    const unsigned cntW = __popc(mask);
    const int leader = __ffs((int)mask) - 1;
    if (lane == leader) waveHist[wv][d] = cntW;
    __syncthreads();
    unsigned before = 0;
    for (int w = 0; w < wv; ++w) before += waveHist[w][d];
    const unsigned pos = offs[(size_t)d*NB_ + blockIdx.x] + runCnt[d] + before + laneRank;
    dst[pos] = rec;
    __syncthreads();
    unsigned add = 0;
    for (int w = 0; w < 8; ++w) add += waveHist[w][ti];
    runCnt[ti] += add;
    __syncthreads();
  }
}

// ---------------------------------------------------- label scan + lovasz ---
__global__ void k_lblsum(const unsigned long long* __restrict__ recs,
                         unsigned* __restrict__ lblBlock) {
  __shared__ unsigned red[256];
  const int ti = threadIdx.x;
  unsigned s = 0;
  for (int j = 0; j < 8; ++j)
    s += (unsigned)(recs[(size_t)blockIdx.x*RTILE + j*256 + ti] & 1ull);
  red[ti] = s; __syncthreads();
  for (int st = 128; st > 0; st >>= 1) {
    if (ti < st) red[ti] += red[ti+st];
    __syncthreads();
  }
  if (ti == 0) lblBlock[blockIdx.x] = red[0];
}

__global__ void k_lblscan(const unsigned* __restrict__ lblBlock,
                          unsigned* __restrict__ lblPrefix) {
  if (threadIdx.x != 0) return;
  unsigned r = 0;
  for (int i = 0; i < NB_; ++i) { const unsigned t = lblBlock[i]; lblPrefix[i] = r; r += t; }
  lblPrefix[NB_] = r;   // total foreground count G
}

__global__ void k_lovasz(const unsigned long long* __restrict__ recs,
                         const unsigned* __restrict__ lblPrefix,
                         float* __restrict__ hdr, int bk) {
  __shared__ unsigned waveSum[8];
  __shared__ unsigned runLbl;
  __shared__ float red[256];
  const int ti = threadIdx.x, lane = ti & 31, wv = ti >> 5;
  if (ti == 0) runLbl = 0;
  __syncthreads();
  const float G = (float)lblPrefix[NB_];
  const unsigned pb = lblPrefix[blockIdx.x];
  float acc = 0.0f;

  for (int j = 0; j < 8; ++j) {
    const int l = j*256 + ti;
    const long long gi = (long long)blockIdx.x*RTILE + l;
    const unsigned long long rec = recs[(size_t)blockIdx.x*RTILE + l];
    const unsigned lbl = (unsigned)(rec & 1ull);
    const unsigned bal = ballot32(lbl != 0u);
    const unsigned inc = __popc(bal & ((2u << lane) - 1u));   // inclusive
    if (lane == 0) waveSum[wv] = __popc(bal);
    __syncthreads();
    unsigned before = 0;
    for (int w = 0; w < wv; ++w) before += waveSum[w];
    const unsigned c = pb + runLbl + before + inc;            // cumsum_i (incl)
    const float fi = (float)gi, ci = (float)c;
    const float jac = 1.0f - (G - ci) / (G + fi + 1.0f - ci);
    float grad;
    if (gi == 0) grad = jac;
    else {
      const float cm = ci - (float)lbl;
      const float jm = 1.0f - (G - cm) / (G + fi - cm);
      grad = jac - jm;
    }
    const unsigned kd = (unsigned)(rec >> 32);
    const unsigned ka = ~kd;
    const unsigned u  = (ka & 0x80000000u) ? (ka & 0x7FFFFFFFu) : ~ka;
    const float err = __uint_as_float(u);
    acc += fmaxf(err, 0.0f) * grad;
    __syncthreads();
    if (ti == 0) { unsigned a = 0; for (int w = 0; w < 8; ++w) a += waveSum[w]; runLbl += a; }
    __syncthreads();
  }
  red[ti] = acc; __syncthreads();
  for (int st = 128; st > 0; st >>= 1) {
    if (ti < st) red[ti] += red[ti+st];
    __syncthreads();
  }
  if (ti == 0) atomicAdd(&hdr[OFF_INSTL + bk], red[0]);
}

// --------------------------------------------------------------- k_final ---
__global__ void k_final(const float* __restrict__ hdr, float* __restrict__ out) {
  if (threadIdx.x != 0) return;
  float tot = 0.0f;
  for (int b = 0; b < B_; ++b) {
    const float im = (hdr[OFF_INSTL + b*K_ + 0] + hdr[OFF_INSTL + b*K_ + 1] +
                      hdr[OFF_INSTL + b*K_ + 2]) * (1.0f / (float)K_);
    const float vl = hdr[OFF_VAR + b] * (1.0f / (float)(K_*T_));
    const float sl = (hdr[OFF_SEEDFG + b] + hdr[OFF_SEEDBG + b]) * (1.0f / (float)HW_);
    tot += 1.0f*im + 10.0f*vl + 1.0f*sl;     // W_INST, W_VAR, W_SEED
  }
  out[0] = tot * (1.0f / (float)B_);
}

// ---------------------------------------------------------------------------
extern "C" void kernel_launch(void* const* d_in, const int* in_sizes, int n_in,
                              void* d_out, int out_size, void* d_ws, size_t ws_size,
                              hipStream_t stream) {
  (void)in_sizes; (void)n_in; (void)out_size; (void)ws_size;
  const float* pred = (const float*)d_in[0];
  const int*   inst = (const int*)d_in[1];
  float* out = (float*)d_out;

  char* base = (char*)d_ws;
  float* hdr = (float*)base;
  unsigned long long* bufA = (unsigned long long*)(base + 4096);
  unsigned long long* bufB = bufA + THW_;
  unsigned* hist      = (unsigned*)(base + 4096 + 2ull * THW_ * 8ull);
  unsigned* offs      = hist + (size_t)256 * NB_;
  unsigned* lblBlock  = offs + (size_t)256 * NB_;
  unsigned* lblPrefix = lblBlock + NB_;

  k_zero<<<(HDR_FLOATS + 255)/256, 256, 0, stream>>>(hdr);
  k_stats<<<B_*T_*100, 256, 0, stream>>>(pred, inst, hdr);
  k_fin1<<<1, 96, 0, stream>>>(hdr);
  k_fin2<<<1, 32, 0, stream>>>(hdr);

  for (int b = 0; b < B_; ++b) {
    for (int k = 0; k < K_; ++k) {
      k_dist<<<THW_/1024, 256, 0, stream>>>(pred, inst, hdr, bufA, b, k, (k == 0) ? 1 : 0);
      unsigned long long* src = bufA;
      unsigned long long* dst = bufB;
      for (int pass = 0; pass < 4; ++pass) {
        const int shift = 32 + 8*pass;
        k_hist<<<NB_, 256, 0, stream>>>(src, hist, shift);
        k_scan<<<1, 256, 0, stream>>>(hist, offs);
        k_scatter<<<NB_, 256, 0, stream>>>(src, dst, offs, shift);
        unsigned long long* tmp = src; src = dst; dst = tmp;
      }
      // after 4 passes sorted data is back in bufA (== src)
      k_lblsum<<<NB_, 256, 0, stream>>>(src, lblBlock);
      k_lblscan<<<1, 32, 0, stream>>>(lblBlock, lblPrefix);
      k_lovasz<<<NB_, 256, 0, stream>>>(src, lblPrefix, hdr, b*K_ + k);
    }
  }
  k_final<<<1, 32, 0, stream>>>(hdr, out);
}
